// KLLoss_678604833494
// MI455X (gfx1250) — matrix operand order repaired and checked by reading, not compiled, not run on hardware
//
#include <hip/hip_runtime.h>
#include <cstdint>

typedef __attribute__((ext_vector_type(16))) _Float16 v16h;
typedef __attribute__((ext_vector_type(8)))  _Float16 v8h;
typedef __attribute__((ext_vector_type(8)))  float    v8f;

#define N_ROWS  4096
#define DDIM    768
#define INV_T   10.0f   // 1/TEMPERATURE
#define CSLICES 16      // column slices per 32-row block
#define MW      32      // rows per wave (two 16-row WMMA tiles)

// ---------------------------------------------------------------------------
// Kernel 1: row-wise L2 normalize (fp32 math), emit f16 copies for WMMA.
// grid = (4096 rows, 4 matrices), block = 256 (3 elements/thread: 768=3*256)
// ---------------------------------------------------------------------------
__global__ __launch_bounds__(256) void normalize_f16_kernel(
    const float* __restrict__ zxs, const float* __restrict__ zys,
    const float* __restrict__ zxt, const float* __restrict__ zyt,
    _Float16* __restrict__ out)
{
    const int row = blockIdx.x;
    const int mat = blockIdx.y;
    const float* src = (mat == 0) ? zxs : (mat == 1) ? zys : (mat == 2) ? zxt : zyt;
    const float* r = src + (size_t)row * DDIM;

    float x0 = r[threadIdx.x];
    float x1 = r[threadIdx.x + 256];
    float x2 = r[threadIdx.x + 512];
    float ss = x0 * x0 + x1 * x1 + x2 * x2;

    __shared__ float red[256];
    red[threadIdx.x] = ss;
    __syncthreads();
    for (int s = 128; s > 0; s >>= 1) {
        if (threadIdx.x < s) red[threadIdx.x] += red[threadIdx.x + s];
        __syncthreads();
    }
    const float norm  = sqrtf(red[0]);
    const float scale = 1.0f / fmaxf(norm, 1e-8f);   // torch cosine-sim eps

    _Float16* o = out + (size_t)mat * N_ROWS * DDIM + (size_t)row * DDIM;
    o[threadIdx.x]       = (_Float16)(x0 * scale);
    o[threadIdx.x + 256] = (_Float16)(x1 * scale);
    o[threadIdx.x + 512] = (_Float16)(x2 * scale);
}

// ---------------------------------------------------------------------------
// Kernel 2: fused dual-GEMM + online softmax statistics.
// Each wave owns a 32-row block (two 16-row A tiles, s+t) and 1/16 of the
// columns (16 col-tiles).  Per K-step: 4 A frags + 2 B frags -> 4 WMMAs
// (B fragments shared by both M tiles).  Per-row stats
//   A=sum(e^a), B=sum(e^b), C=sum(e^b(b-a)), D=sum(e^a(a-b))
// accumulate in registers and land in a per-slice partial array (no atomics).
// grid = 256 blocks x 8 waves = 2048 waves.
// ---------------------------------------------------------------------------
__global__ __launch_bounds__(256) void sim_kl_rows_kernel(
    const _Float16* __restrict__ nrm, float* __restrict__ partial)
{
    const int lane     = threadIdx.x & 31;
    const int wave     = threadIdx.x >> 5;
    const int flat     = blockIdx.x * 8 + wave;        // 0..2047
    const int rowBlock = flat / CSLICES;               // 0..127
    const int slice    = flat % CSLICES;               // 0..15
    const int rowBase  = rowBlock * MW;
    const int tilesPerSlice = (N_ROWS / 16) / CSLICES; // 16
    const int colTile0 = slice * tilesPerSlice;

    const _Float16* xs = nrm;                                 // zxs_n
    const _Float16* ys = nrm + (size_t)1 * N_ROWS * DDIM;     // zys_n
    const _Float16* xt = nrm + (size_t)2 * N_ROWS * DDIM;     // zxt_n
    const _Float16* yt = nrm + (size_t)3 * N_ROWS * DDIM;     // zyt_n

    const int m  = lane & 15;   // A: tile row / B: tile col owned by this lane
    const int kh = lane >> 4;   // K-half selector (ISA 16-bit A/B layouts)

    const _Float16* aS0 = xs + (size_t)(rowBase + m) * DDIM;
    const _Float16* aS1 = xs + (size_t)(rowBase + 16 + m) * DDIM;
    const _Float16* aT0 = xt + (size_t)(rowBase + m) * DDIM;
    const _Float16* aT1 = xt + (size_t)(rowBase + 16 + m) * DDIM;

    // [mtile*8 + r]
    float accA[16], accB[16], accC[16], accD[16];
#pragma unroll
    for (int r = 0; r < 16; ++r) { accA[r] = 0.f; accB[r] = 0.f; accC[r] = 0.f; accD[r] = 0.f; }

    for (int jt = 0; jt < tilesPerSlice; ++jt) {
        const int colBase = (colTile0 + jt) * 16;
        const _Float16* bS = ys + (size_t)(colBase + m) * DDIM;
        const _Float16* bT = yt + (size_t)(colBase + m) * DDIM;

        v8f cs0 = {}, cs1 = {}, ct0 = {}, ct1 = {};
#pragma unroll 2
        for (int k0 = 0; k0 < DDIM; k0 += 32) {
            // A (16x32 f16): lane kh=0 -> K {0..7,16..23}, kh=1 -> K {8..15,24..31}
            v8h s0lo = *(const v8h*)(aS0 + k0 + kh * 8);
            v8h s0hi = *(const v8h*)(aS0 + k0 + 16 + kh * 8);
            v8h s1lo = *(const v8h*)(aS1 + k0 + kh * 8);
            v8h s1hi = *(const v8h*)(aS1 + k0 + 16 + kh * 8);
            v8h t0lo = *(const v8h*)(aT0 + k0 + kh * 8);
            v8h t0hi = *(const v8h*)(aT0 + k0 + 16 + kh * 8);
            v8h t1lo = *(const v8h*)(aT1 + k0 + kh * 8);
            v8h t1hi = *(const v8h*)(aT1 + k0 + 16 + kh * 8);
            v16h as0, as1, at0, at1;
#pragma unroll
            for (int i = 0; i < 8; ++i) {
                as0[i] = s0lo[i]; as0[i + 8] = s0hi[i];
                as1[i] = s1lo[i]; as1[i + 8] = s1hi[i];
                at0[i] = t0lo[i]; at0[i + 8] = t0hi[i];
                at1[i] = t1lo[i]; at1[i + 8] = t1hi[i];
            }
            // B (32x16 f16): lane holds col N=m, same K order as A: kh=0 -> K {0..7,16..23}, kh=1 -> K {8..15,24..31}
            v16h bs = __builtin_shufflevector(*(const v8h*)(bS + k0 + kh * 8), *(const v8h*)(bS + k0 + 16 + kh * 8), 0, 1, 2, 3, 4, 5, 6, 7, 8, 9, 10, 11, 12, 13, 14, 15);
            v16h bt = __builtin_shufflevector(*(const v8h*)(bT + k0 + kh * 8), *(const v8h*)(bT + k0 + 16 + kh * 8), 0, 1, 2, 3, 4, 5, 6, 7, 8, 9, 10, 11, 12, 13, 14, 15);

            cs0 = __builtin_amdgcn_wmma_f32_16x16x32_f16(false, as0, false, bs,
                                                         (short)0, cs0, false, false);
            cs1 = __builtin_amdgcn_wmma_f32_16x16x32_f16(false, as1, false, bs,
                                                         (short)0, cs1, false, false);
            ct0 = __builtin_amdgcn_wmma_f32_16x16x32_f16(false, at0, false, bt,
                                                         (short)0, ct0, false, false);
            ct1 = __builtin_amdgcn_wmma_f32_16x16x32_f16(false, at1, false, bt,
                                                         (short)0, ct1, false, false);
        }
        // C/D layout: VGPR r, lanes 0-15 -> (M=r, N=lane); lanes 16-31 -> (M=r+8)
#pragma unroll
        for (int r = 0; r < 8; ++r) {
            {
                float a  = cs0[r] * INV_T;
                float b  = ct0[r] * INV_T;
                float ea = __expf(a);
                float eb = __expf(b);
                accA[r] += ea; accB[r] += eb;
                accC[r] += eb * (b - a); accD[r] += ea * (a - b);
            }
            {
                float a  = cs1[r] * INV_T;
                float b  = ct1[r] * INV_T;
                float ea = __expf(a);
                float eb = __expf(b);
                accA[8 + r] += ea; accB[8 + r] += eb;
                accC[8 + r] += eb * (b - a); accD[8 + r] += ea * (a - b);
            }
        }
    }

    // Reduce over the 16 lanes holding the same output row (masks 1..8 never
    // cross the 16-lane halves, so rows r and r+8 reduce independently).
    float* base = partial + (size_t)slice * 4 * N_ROWS;
#pragma unroll
    for (int t = 0; t < 2; ++t) {
#pragma unroll
        for (int r = 0; r < 8; ++r) {
            float A = accA[t * 8 + r], B = accB[t * 8 + r];
            float C = accC[t * 8 + r], D = accD[t * 8 + r];
            for (int s = 1; s < 16; s <<= 1) {
                A += __shfl_xor(A, s, 32);
                B += __shfl_xor(B, s, 32);
                C += __shfl_xor(C, s, 32);
                D += __shfl_xor(D, s, 32);
            }
            if ((lane & 15) == 0) {
                const int row = rowBase + t * 16 + r + 8 * kh;
                base[row]              = A;
                base[N_ROWS + row]     = B;
                base[2 * N_ROWS + row] = C;
                base[3 * N_ROWS + row] = D;
            }
        }
    }
}

// ---------------------------------------------------------------------------
// Kernel 3: fold CSLICES partials per row, then reduce 4096 rows -> 2 losses.
// ---------------------------------------------------------------------------
__global__ __launch_bounds__(256) void finalize_kernel(
    const float* __restrict__ partial, float* __restrict__ out)
{
    float ls = 0.f, lt = 0.f;
    for (int i = threadIdx.x; i < N_ROWS; i += 256) {
        float A = 0.f, B = 0.f, C = 0.f, D = 0.f;
        for (int s = 0; s < CSLICES; ++s) {
            const float* base = partial + (size_t)s * 4 * N_ROWS;
            A += base[i];
            B += base[N_ROWS + i];
            C += base[2 * N_ROWS + i];
            D += base[3 * N_ROWS + i];
        }
        float logA = __logf(A), logB = __logf(B);
        ls += C / B + logA - logB;
        lt += D / A + logB - logA;
    }
    __shared__ float r0[256], r1[256];
    r0[threadIdx.x] = ls;
    r1[threadIdx.x] = lt;
    __syncthreads();
    for (int s = 128; s > 0; s >>= 1) {
        if (threadIdx.x < s) {
            r0[threadIdx.x] += r0[threadIdx.x + s];
            r1[threadIdx.x] += r1[threadIdx.x + s];
        }
        __syncthreads();
    }
    if (threadIdx.x == 0) {
        const float inv = 1.0f / ((float)N_ROWS * (float)N_ROWS);
        out[0] = r0[0] * inv;   // loss_s
        out[1] = r1[0] * inv;   // loss_t
    }
}

extern "C" void kernel_launch(void* const* d_in, const int* in_sizes, int n_in,
                              void* d_out, int out_size, void* d_ws, size_t ws_size,
                              hipStream_t stream) {
    (void)in_sizes; (void)n_in; (void)out_size; (void)ws_size;
    const float* zxs = (const float*)d_in[0];
    const float* zys = (const float*)d_in[1];
    const float* zxt = (const float*)d_in[2];
    const float* zyt = (const float*)d_in[3];

    _Float16* nrm     = (_Float16*)d_ws;                       // 4 * 4096*768 f16 = 25 MB
    float*    partial = (float*)((char*)d_ws +
                          (size_t)4 * N_ROWS * DDIM * sizeof(_Float16)); // 16*4*4096 f32 = 1 MB

    dim3 g1(N_ROWS, 4);
    normalize_f16_kernel<<<g1, 256, 0, stream>>>(zxs, zys, zxt, zyt, nrm);

    const int totalWaves = (N_ROWS / MW) * CSLICES;            // 2048
    sim_kl_rows_kernel<<<totalWaves / 8, 256, 0, stream>>>(nrm, partial);

    finalize_kernel<<<1, 256, 0, stream>>>(partial, (float*)d_out);
}
